// MixtureDecoder_72275709657592
// MI455X (gfx1250) — matrix operand order repaired and checked by reading, not compiled, not run on hardware
//
#include <hip/hip_runtime.h>
#include <hip/hip_bf16.h>
#include <math.h>

// ---------- types ----------
typedef __attribute__((ext_vector_type(16))) __bf16        v16bf;
typedef __attribute__((ext_vector_type(8)))  float         v8f;
typedef __attribute__((ext_vector_type(4)))  unsigned int  v4u;

union Frag32B { v4u q[2]; v16bf v; };

__device__ __forceinline__ unsigned short f2bf(float f) {
  unsigned int u = __float_as_uint(f);
  unsigned int r = (u + 0x7FFFu + ((u >> 16) & 1u)) >> 16;   // RNE
  return (unsigned short)r;
}

#define BATCH   512
#define LATENT  128
#define GATES   31
#define LEAF    32

// ---------- 0) zero-fill the boundary zero-page (ws is poisoned) ----------
__global__ __launch_bounds__(256) void zerobuf_kernel(unsigned int* __restrict__ z) {
  z[blockIdx.x * 256 + threadIdx.x] = 0u;   // 2 blocks x 256 x 4B = 2KB
}

// ---------- 1) gating + soft-tree leaf probabilities (bf16 out, one wave/row) ----------
__global__ __launch_bounds__(256) void gate_leaf_kernel(
    const float* __restrict__ x, const float* __restrict__ gw,
    const float* __restrict__ gb, unsigned short* __restrict__ lpbf)
{
  const int lane = threadIdx.x & 31;
  const int row  = blockIdx.x * 8 + (threadIdx.x >> 5);
  const float* xr = x + (size_t)row * LATENT;
  const int gi = lane < GATES ? lane : 0;
  float t = 0.f;
  #pragma unroll 8
  for (int k = 0; k < LATENT; ++k) t += xr[k] * gw[k * GATES + gi];
  t += gb[gi];
  const float g = 1.0f / (1.0f + expf(-t));
  float prob = 1.0f;
  const int L = lane;
  #pragma unroll
  for (int d = 1; d <= 5; ++d) {
    const int pd  = L >> (6 - d);
    const int off = (1 << (d - 1)) - 1;
    const int bd  = (L >> (5 - d)) & 1;
    const float gd = __shfl(g, off + pd, 32);
    prob *= bd ? (1.0f - gd) : gd;
  }
  lpbf[(size_t)row * LEAF + lane] = f2bf(prob);
}

// ---------- 2a) z (f32 [8192,32]) -> bf16 ----------
__global__ __launch_bounds__(256) void z2bf_kernel(
    const float* __restrict__ z, unsigned short* __restrict__ zbf) {
  const int idx = blockIdx.x * 256 + threadIdx.x;
  if (idx < 8192 * LEAF) zbf[idx] = f2bf(z[idx]);
}

// ---------- 2b) act0[n,yx,c] = sum_l lp[n,l] * z[c*64+yx, l]  (1 WMMA / tile) ----------
// Tile: M=16 batch rows, N=16 channels at fixed yx, K=32 leaves.
__global__ __launch_bounds__(256) void leafmix_wmma_kernel(
    const unsigned short* __restrict__ lpbf,  // [512,32] bf16
    const unsigned short* __restrict__ zbf,   // [8192,32] bf16
    unsigned short* __restrict__ act0)        // [512,8,8,128] bf16
{
  const int lane = threadIdx.x & 31;
  const int wave = threadIdx.x >> 5;
  const int l15 = lane & 15, half = lane >> 4;
  const int w = blockIdx.x * 8 + wave;        // 32*8*64 = 16384 tiles
  const int btile = w & 31;
  const int rest  = w >> 5;
  const int ctile = rest & 7;
  const int yx    = rest >> 3;

  // A = lp rows (m = batch): lane K-offset 8*half elements = 16 bytes
  const char* Ar = (const char*)(lpbf + (size_t)(btile * 16 + l15) * LEAF) + 16 * half;
  Frag32B af;
  af.q[0] = *(const v4u*)(Ar);
  af.q[1] = *(const v4u*)(Ar + 32);
  // B = z rows as columns (n = channel): lane K-offset 8*half elements = 16 bytes
  const char* Br = (const char*)(zbf + (size_t)((ctile * 16 + l15) * 64 + yx) * LEAF) + 16 * half;
  Frag32B bf;
  bf.q[0] = *(const v4u*)(Br);
  bf.q[1] = *(const v4u*)(Br + 32);

  v8f acc = {0.f,0.f,0.f,0.f,0.f,0.f,0.f,0.f};
  acc = __builtin_amdgcn_wmma_f32_16x16x32_bf16(
      false, af.v, false, bf.v, (short)0, acc, false, false);

  // D: lane = channel column, VGPR v -> batch = btile*16 + v + 8*half
  #pragma unroll
  for (int v = 0; v < 8; ++v) {
    const int bat = btile * 16 + v + 8 * half;
    act0[(size_t)(bat * 64 + yx) * 128 + ctile * 16 + l15] = f2bf(acc[v]);
  }
}

// ---------- 3) weight prep: Abuf[p][co][t*CIN+ci] = bf16(w[ci][co][ky][kx]) ----------
__global__ __launch_bounds__(256) void prep_w_kernel(
    const float* __restrict__ w, unsigned short* __restrict__ Abuf,
    int CIN, int COUT, int COUTP)
{
  const int idx = blockIdx.x * 256 + threadIdx.x;
  const int total = 4 * COUTP * 4 * CIN;
  if (idx >= total) return;
  const int ci = idx % CIN;
  const int t  = (idx / CIN) & 3;
  const int co = (idx / (CIN * 4)) % COUTP;
  const int p  = idx / (CIN * 4 * COUTP);
  const int ty = t >> 1, tx = t & 1, py = p >> 1, px = p & 1;
  const int ky = (py == 0) ? (ty ? 3 : 1) : (ty ? 2 : 0);
  const int kx = (px == 0) ? (tx ? 3 : 1) : (tx ? 2 : 0);
  float v = (co < COUT) ? w[(((size_t)ci * COUT + co) * 4 + ky) * 4 + kx] : 0.0f;
  Abuf[idx] = f2bf(v);
}

// ---------- 4) ConvTranspose2d(k4,s2,p1) implicit GEMM, MT x NT tiles per wave ----------
template<int CIN, int COUT, int COUTP, int HI, int WI, int MT, int NT, int MW, int NW,
         bool FINAL>
__global__ __launch_bounds__(256) void convt_wmma_kernel(
    const unsigned short* __restrict__ in,    // [B, HI, WI, CIN] bf16
    const unsigned short* __restrict__ Abuf,  // [4][COUTP][4*CIN] bf16
    const unsigned short* __restrict__ zbuf,  // >= 2KB zeros
    const float* __restrict__ bias,
    void* __restrict__ outp)
{
  constexpr int K  = 4 * CIN;
  constexpr int HO = 2 * HI, WO = 2 * WI;
  constexpr int PIX_PER_WAVE  = NT * 16;
  constexpr int PIX_PER_BLOCK = NW * PIX_PER_WAVE;
  static_assert(MW * NW == 8, "8 waves per block");
  static_assert(MT * 16 * MW == COUTP, "M tiling covers COUTP");

  const int lane = threadIdx.x & 31;
  const int wave = threadIdx.x >> 5;
  const int mw = wave % MW;
  const int nw = wave / MW;
  const int p  = blockIdx.y;
  const int py = p >> 1, px = p & 1;
  const int l15  = lane & 15;
  const int half = lane >> 4;
  const int hoffA = 16 * half;   // bytes: A lane K-offset = 8*half elements
  const int hoffB = 16 * half;   // bytes: B lane K-offset = 8*half elements

  // pixel decode per n-tile
  int an[NT], bn[NT], nn[NT];
  #pragma unroll
  for (int ni = 0; ni < NT; ++ni) {
    const int u = blockIdx.x * PIX_PER_BLOCK + nw * PIX_PER_WAVE + ni * 16 + l15;
    bn[ni] = u % WI; an[ni] = (u / WI) % HI; nn[ni] = u / (WI * HI);
  }

  // A base pointers per m-tile
  const char* Ab[MT];
  #pragma unroll
  for (int mi = 0; mi < MT; ++mi) {
    const int co = mw * (MT * 16) + mi * 16 + l15;
    Ab[mi] = (const char*)(Abuf + (size_t)(p * COUTP + co) * K) + hoffA;
  }

  v8f acc[MT][NT];
  #pragma unroll
  for (int mi = 0; mi < MT; ++mi)
    #pragma unroll
    for (int ni = 0; ni < NT; ++ni)
      acc[mi][ni] = (v8f){0.f,0.f,0.f,0.f,0.f,0.f,0.f,0.f};

  #pragma unroll
  for (int t = 0; t < 4; ++t) {
    const int ty = t >> 1, tx = t & 1;
    const int dy = (py == 0) ? (ty ? -1 : 0) : (ty ? 0 : 1);
    const int dx = (px == 0) ? (tx ? -1 : 0) : (tx ? 0 : 1);

    const char* Bb[NT];
    #pragma unroll
    for (int ni = 0; ni < NT; ++ni) {
      const int iy = an[ni] + dy, ix = bn[ni] + dx;
      const bool ok = (iy >= 0) & (iy < HI) & (ix >= 0) & (ix < WI);
      Bb[ni] = (const char*)(ok
          ? in + (size_t)((nn[ni] * HI + iy) * WI + ix) * CIN : zbuf) + hoffB;
    }

    #pragma unroll
    for (int kk = 0; kk < CIN / 32; ++kk) {
      const int k0 = t * CIN + kk * 32;   // compile-time
      Frag32B bf[NT];
      #pragma unroll
      for (int ni = 0; ni < NT; ++ni) {
        bf[ni].q[0] = *(const v4u*)(Bb[ni] + kk * 64);
        bf[ni].q[1] = *(const v4u*)(Bb[ni] + kk * 64 + 32);
      }
      Frag32B af[MT];
      #pragma unroll
      for (int mi = 0; mi < MT; ++mi) {
        af[mi].q[0] = *(const v4u*)(Ab[mi] + k0 * 2);
        af[mi].q[1] = *(const v4u*)(Ab[mi] + k0 * 2 + 32);
      }
      #pragma unroll
      for (int mi = 0; mi < MT; ++mi)
        #pragma unroll
        for (int ni = 0; ni < NT; ++ni)
          acc[mi][ni] = __builtin_amdgcn_wmma_f32_16x16x32_bf16(
              false, af[mi].v, false, bf[ni].v, (short)0, acc[mi][ni], false, false);
    }
  }

  if constexpr (!FINAL) {
    unsigned short* out = (unsigned short*)outp;
    #pragma unroll
    for (int ni = 0; ni < NT; ++ni) {
      const int oy = 2 * an[ni] + py, ox = 2 * bn[ni] + px;
      const size_t pixoff = (size_t)((nn[ni] * HO + oy) * WO + ox) * COUT;
      #pragma unroll
      for (int mi = 0; mi < MT; ++mi) {
        const int cb = mw * (MT * 16) + mi * 16 + 8 * half;
        union { unsigned short h[8]; v4u q; } pk;
        #pragma unroll
        for (int v = 0; v < 8; ++v) {
          float f = acc[mi][ni][v] + bias[cb + v];
          pk.h[v] = f2bf(f > 0.f ? f : 0.f);
        }
        *(v4u*)(out + pixoff + cb) = pk.q;
      }
    }
  } else {
    // fp32 NCHW, channels 0..2 real (COUT padded to 16); MT must be 1
    if (half == 0) {
      float* out = (float*)outp;
      #pragma unroll
      for (int ni = 0; ni < NT; ++ni) {
        const int oy = 2 * an[ni] + py, ox = 2 * bn[ni] + px;
        #pragma unroll
        for (int v = 0; v < 3; ++v)
          out[((size_t)(nn[ni] * 3 + v) * HO + oy) * WO + ox] = acc[0][ni][v] + bias[v];
      }
    }
  }
}

// ---------- launch ----------
extern "C" void kernel_launch(void* const* d_in, const int* in_sizes, int n_in,
                              void* d_out, int out_size, void* d_ws, size_t ws_size,
                              hipStream_t stream) {
  (void)in_sizes; (void)n_in; (void)out_size; (void)ws_size;
  const float* x  = (const float*)d_in[0];
  const float* gw = (const float*)d_in[1];
  const float* gb = (const float*)d_in[2];
  const float* z  = (const float*)d_in[3];
  const float* w1 = (const float*)d_in[4];
  const float* b1 = (const float*)d_in[5];
  const float* w2 = (const float*)d_in[6];
  const float* b2 = (const float*)d_in[7];
  const float* w3 = (const float*)d_in[8];
  const float* b3 = (const float*)d_in[9];

  char* ws = (char*)d_ws;
  unsigned short* lpbf = (unsigned short*)(ws + 0);              //  32 KB [512,32]
  unsigned short* A1   = (unsigned short*)(ws + (64u << 10));    // 512 KB
  unsigned short* A2   = (unsigned short*)(ws + (576u << 10));   // 256 KB
  unsigned short* A3   = (unsigned short*)(ws + (832u << 10));   //  32 KB
  unsigned short* zb   = (unsigned short*)(ws + (864u << 10));   //   2 KB zeros
  unsigned short* zbf  = (unsigned short*)(ws + (1u << 20));     // 512 KB [8192,32]
  unsigned short* act0 = (unsigned short*)(ws + (2u << 20));     //   8 MB [512, 8, 8,128]
  unsigned short* act1 = (unsigned short*)(ws + (10u << 20));    //  32 MB [512,16,16,128]
  unsigned short* act2 = (unsigned short*)(ws + (42u << 20));    //  64 MB [512,32,32, 64]
  float* out = (float*)d_out;                                    // [512,3,64,64]

  zerobuf_kernel<<<2, 256, 0, stream>>>((unsigned int*)zb);
  gate_leaf_kernel<<<BATCH / 8, 256, 0, stream>>>(x, gw, gb, lpbf);
  z2bf_kernel<<<(8192 * LEAF) / 256, 256, 0, stream>>>(z, zbf);

  prep_w_kernel<<<(4 * 128 * 4 * 128 + 255) / 256, 256, 0, stream>>>(w1, A1, 128, 128, 128);
  prep_w_kernel<<<(4 *  64 * 4 * 128 + 255) / 256, 256, 0, stream>>>(w2, A2, 128,  64,  64);
  prep_w_kernel<<<(4 *  16 * 4 *  64 + 255) / 256, 256, 0, stream>>>(w3, A3,  64,   3,  16);

  // leaf mixture GEMM: 16384 single-WMMA tiles, 8 waves/block
  leafmix_wmma_kernel<<<2048, 256, 0, stream>>>(lpbf, zbf, act0);

  // conv1: 128->128, 8x8 -> 16x16, ReLU. 2x2 tiles/wave, MW=4, NW=2 -> 64 pix/block
  convt_wmma_kernel<128, 128, 128, 8, 8, 2, 2, 4, 2, false>
      <<<dim3(512, 4), 256, 0, stream>>>(act0, A1, zb, b1, act1);

  // conv2: 128->64, 16x16 -> 32x32, ReLU. 2x2 tiles/wave, MW=2, NW=4 -> 128 pix/block
  convt_wmma_kernel<128, 64, 64, 16, 16, 2, 2, 2, 4, false>
      <<<dim3(1024, 4), 256, 0, stream>>>(act1, A2, zb, b2, act2);

  // conv3: 64->3 (pad 16), 32x32 -> 64x64, fp32 NCHW. 1x4 tiles/wave, NW=8 -> 512 pix/block
  convt_wmma_kernel<64, 3, 16, 32, 32, 1, 4, 1, 8, true>
      <<<dim3(1024, 4), 256, 0, stream>>>(act2, A3, zb, b3, out);
}
